// aggregate_to_node_6828998001288
// MI455X (gfx1250) — compile-verified, matrix operand order checked by reading, not run on hardware
//
#include <hip/hip_runtime.h>
#include <hip/hip_bf16.h>
#include <stdint.h>

// ---------------------------------------------------------------------------
// CDNA5 / gfx1250 fused edge-MLP + scatter-add.
// All GEMM work on v_wmma_f32_16x16x32_bf16 (f32 accumulate).
// ---------------------------------------------------------------------------

typedef __attribute__((ext_vector_type(16))) __bf16 v16bf;
typedef __attribute__((ext_vector_type(8)))  float  v8f;

union Frag {
    v16bf v;
    uint4 q[2];   // 32 bytes = 16 bf16
};

#define HID 128
#define EMB 128
#define WMAT (128 * 128)
#define TILE_M 128          // edges per block
#define LDS_COLS 136        // 128 + 8 bf16 pad -> 272B row stride (bank spread)

__device__ __forceinline__ unsigned short f32_to_bf16(float f) {
    uint32_t u = __builtin_bit_cast(uint32_t, f);
    // round-to-nearest-even truncation to bf16
    u += 0x7FFFu + ((u >> 16) & 1u);
    return (unsigned short)(u >> 16);
}

// Pack two f32 -> two bf16 in one dword (v_cvt_pk_bf16_f32 when available).
__device__ __forceinline__ unsigned int pack2_bf16(float a, float b) {
#if defined(__AMDGCN__) && __has_builtin(__builtin_amdgcn_cvt_pk_bf16_f32)
    auto pk = __builtin_amdgcn_cvt_pk_bf16_f32(a, b);
    static_assert(sizeof(pk) == 4, "expected 2xbf16 in one dword");
    return __builtin_bit_cast(unsigned int, pk);
#else
    return (unsigned int)f32_to_bf16(a) | ((unsigned int)f32_to_bf16(b) << 16);
#endif
}

__device__ __forceinline__ float fast_swish(float x) {
    // x * sigmoid(x); raw v_rcp_f32 (no Newton iteration) is plenty for an
    // activation feeding a bf16 store.
    return x * __builtin_amdgcn_rcpf(1.0f + __expf(-x));
}

// ---------------------------------------------------------------------------
// Prep: zero output accumulator; convert+transpose weights to bf16 in d_ws.
//   ws layout (bf16):
//     [0*WMAT .. 4*WMAT) : Wt[layer][n][k]  (n = out col, k = in col), layers 0..3
//     [4*WMAT .. 4*WMAT + 16*128) : Wt_out[16][128], rows 3..15 zero
// ---------------------------------------------------------------------------
__global__ void prep_kernel(const float* __restrict__ W_in,
                            const float* __restrict__ Ws,
                            const float* __restrict__ W_out,
                            unsigned short* __restrict__ wt,
                            float* __restrict__ out,
                            int out_elems, int prep_elems) {
    int tid = blockIdx.x * blockDim.x + threadIdx.x;
    if (tid < out_elems) out[tid] = 0.0f;
    if (tid < 4 * WMAT) {
        int m = tid >> 14;            // which 128x128 matrix
        int idx = tid & (WMAT - 1);
        int n = idx >> 7;             // output column (row of transposed)
        int k = idx & 127;            // input column
        float v = (m == 0) ? W_in[k * 128 + n]
                           : Ws[(m - 1) * WMAT + k * 128 + n];
        wt[m * WMAT + n * 128 + k] = f32_to_bf16(v);
    } else if (tid < prep_elems) {
        int idx = tid - 4 * WMAT;     // 16*128 tail
        int n = idx >> 7;
        int k = idx & 127;
        float v = (n < 3) ? W_out[k * 3 + n] : 0.0f;
        wt[4 * WMAT + n * 128 + k] = f32_to_bf16(v);
    }
}

// ---------------------------------------------------------------------------
// Fused MLP + direction scale + scatter-add. 256 threads = 8 waves.
// Each block: 128 edges. Each wave: 16-row strip, all 128 columns.
// ---------------------------------------------------------------------------
__global__ __launch_bounds__(256) void fused_mlp_scatter(
        const float* __restrict__ e,
        const int*   __restrict__ nidx,
        const float* __restrict__ dist_vec,
        const float* __restrict__ b_in,
        const float* __restrict__ bs,
        const unsigned short* __restrict__ wt,
        float* __restrict__ out,
        int E) {
    __shared__ unsigned short hs[TILE_M * LDS_COLS];   // 34816 B

    const int tid  = threadIdx.x;
    const int lane = tid & 31;
    const int wave = tid >> 5;
    const long e0  = (long)blockIdx.x * TILE_M;

    // ---- stage e tile -> LDS as bf16 (row-major, padded stride) ----
    // 128x128 floats = 4096 float4; 16 float4 per thread.
#pragma unroll
    for (int it = 0; it < 16; ++it) {
        int f  = tid + 256 * it;        // float4 index in tile
        int r  = f >> 5;                // 32 float4 per row
        int c4 = f & 31;
        long row = e0 + r;
        uint2 pk;
        if (row < (long)E) {
            float4 d = ((const float4*)(e + row * HID))[c4];
            pk.x = pack2_bf16(d.x, d.y);
            pk.y = pack2_bf16(d.z, d.w);
        } else {
            pk.x = 0u; pk.y = 0u;
        }
        *(uint2*)&hs[r * LDS_COLS + c4 * 4] = pk;
    }
    __syncthreads();

    const int m0      = wave * 16;            // this wave's row strip
    const int krowoff = (lane >> 4) << 3;     // 0 or 8 (A frag K split / C row split)
    const int l15     = lane & 15;
    const int arow    = m0 + l15;

    // ---- 4 dense 128x128 layers ----
    for (int layer = 0; layer < 4; ++layer) {
        const unsigned short* Wt   = wt + layer * WMAT;
        const float*          bias = (layer == 0) ? b_in : (bs + (layer - 1) * EMB);

        v8f acc[8];
#pragma unroll
        for (int t = 0; t < 8; ++t) acc[t] = {};

#pragma unroll
        for (int k = 0; k < 4; ++k) {
            const int kb = k * 32;
            Frag a;
            const unsigned short* ap = &hs[arow * LDS_COLS + kb + krowoff];
            a.q[0] = *(const uint4*)ap;
            a.q[1] = *(const uint4*)(ap + 16);
#pragma unroll
            for (int t = 0; t < 8; ++t) {
                Frag b;
                const unsigned short* bp = Wt + (t * 16 + l15) * 128 + kb + krowoff;
                b.q[0] = *(const uint4*)bp;
                b.q[1] = *(const uint4*)(bp + 16);
                acc[t] = __builtin_amdgcn_wmma_f32_16x16x32_bf16(
                    false, a.v, false, b.v, (short)0, acc[t], false, false);
            }
        }
        __syncthreads();   // everyone done reading hs for this layer

        // bias + (swish for layers 1..3) + write back bf16
        const int mbase = m0 + krowoff;
#pragma unroll
        for (int t = 0; t < 8; ++t) {
            const int n = t * 16 + l15;
            const float bv = bias[n];
#pragma unroll
            for (int i2 = 0; i2 < 8; i2 += 2) {
                float x0 = acc[t][i2]     + bv;
                float x1 = acc[t][i2 + 1] + bv;
                if (layer > 0) { x0 = fast_swish(x0); x1 = fast_swish(x1); }
                unsigned int p = pack2_bf16(x0, x1);
                hs[(mbase + i2)     * LDS_COLS + n] = (unsigned short)p;
                hs[(mbase + i2 + 1) * LDS_COLS + n] = (unsigned short)(p >> 16);
            }
        }
        __syncthreads();
    }

    // ---- final 128 -> 3 projection via one WMMA n-tile (padded W_out^T) ----
    {
        const unsigned short* Wt = wt + 4 * WMAT;
        v8f acc = {};
#pragma unroll
        for (int k = 0; k < 4; ++k) {
            const int kb = k * 32;
            Frag a, b;
            const unsigned short* ap = &hs[arow * LDS_COLS + kb + krowoff];
            a.q[0] = *(const uint4*)ap;
            a.q[1] = *(const uint4*)(ap + 16);
            const unsigned short* bp = Wt + l15 * 128 + kb + krowoff;
            b.q[0] = *(const uint4*)bp;
            b.q[1] = *(const uint4*)(bp + 16);
            acc = __builtin_amdgcn_wmma_f32_16x16x32_bf16(
                false, a.v, false, b.v, (short)0, acc, false, false);
        }

        const int n = l15;          // output channel this lane holds
        if (n < 3) {
            const int mbase = m0 + krowoff;
#pragma unroll
            for (int i2 = 0; i2 < 8; ++i2) {
                long row = e0 + mbase + i2;
                if (row < (long)E) {
                    float dx = dist_vec[row * 3 + 0];
                    float dy = dist_vec[row * 3 + 1];
                    float dz = dist_vec[row * 3 + 2];
                    float inv = rsqrtf(dx * dx + dy * dy + dz * dz);
                    float u = dist_vec[row * 3 + n] * inv;
                    float val = acc[i2] * u;
                    atomicAdd(&out[(long)nidx[row] * 3 + n], val);
                }
            }
        }
    }
}

// ---------------------------------------------------------------------------
// Host launcher.
// Input order: e, i, dist_vec, W_in, b_in, Ws, bs, W_out, node_num
// ---------------------------------------------------------------------------
extern "C" void kernel_launch(void* const* d_in, const int* in_sizes, int n_in,
                              void* d_out, int out_size, void* d_ws, size_t ws_size,
                              hipStream_t stream) {
    const float* e        = (const float*)d_in[0];
    const int*   nidx     = (const int*)  d_in[1];
    const float* dist_vec = (const float*)d_in[2];
    const float* W_in     = (const float*)d_in[3];
    const float* b_in     = (const float*)d_in[4];
    const float* Ws       = (const float*)d_in[5];
    const float* bs       = (const float*)d_in[6];
    const float* W_out    = (const float*)d_in[7];

    const int E = in_sizes[1];                 // edge count (index array length)
    unsigned short* wt = (unsigned short*)d_ws; // bf16 weights: 4*16384 + 16*128
    float* out = (float*)d_out;

    const int prep_elems = 4 * WMAT + 16 * 128;      // 67584
    int total = prep_elems > out_size ? prep_elems : out_size;
    prep_kernel<<<(total + 255) / 256, 256, 0, stream>>>(
        W_in, Ws, W_out, wt, out, out_size, prep_elems);

    const int grid = (E + TILE_M - 1) / TILE_M;      // 5000 blocks
    fused_mlp_scatter<<<grid, 256, 0, stream>>>(
        e, nidx, dist_vec, b_in, bs, wt, out, E);
}
